// MultiHeadAttention_53163105190136
// MI455X (gfx1250) — matrix operand order repaired and checked by reading, not compiled, not run on hardware
//
#include <hip/hip_runtime.h>
#include <hip/hip_bf16.h>

// ---------------------------------------------------------------------------
// MHA forward for MI455X (gfx1250, wave32, WMMA bf16).
//   out = softmax_causal((xWq)(xWk)^T / sqrt(hd)) (xWv) Wo^T
// Pipeline: fp32->bf16 convert, 3x WMMA GEMM (QKV), flash attention (S^T
// trick so P is already in A-fragment layout), final WMMA GEMM (fp32 out).
// This revision: GEMM tile staging uses CDNA5 async global->LDS DMA
// (global_load_async_to_lds_b128 + s_wait_asynccnt), double-buffered with
// one barrier per K-step. Attention unchanged (schedule already overlaps
// loads/exp/bpermute under the WMMA stream).
// ---------------------------------------------------------------------------

typedef __attribute__((ext_vector_type(16))) __bf16 v16bf;
typedef __attribute__((ext_vector_type(8)))  __bf16 v8bf;
typedef __attribute__((ext_vector_type(4)))  __bf16 v4bf;
typedef __attribute__((ext_vector_type(8)))  float  v8f;

#define BATCH 4
#define SEQ   2048
#define DM    2048          // d_in == d_out
#define NH    16
#define HD    128
#define MTOT  (BATCH * SEQ) // 8192 rows

static __device__ __forceinline__ v16bf cat8(v8bf lo, v8bf hi) {
  v16bf r;
#pragma unroll
  for (int i = 0; i < 8; ++i) { r[i] = lo[i]; r[i + 8] = hi[i]; }
  return r;
}

static __device__ __forceinline__ v8f wmma_bf16(v16bf a, v16bf b, v8f c) {
  // D = A(16x32 bf16) * B(32x16 bf16) + C(16x16 f32)
  return __builtin_amdgcn_wmma_f32_16x16x32_bf16(false, a, false, b,
                                                 (short)0, c, false, false);
}

// CDNA5 async DMA: global memory -> LDS, 16 bytes per lane, ASYNCcnt-tracked.
// LDS byte address comes from the first VGPR operand, global address from the second.
static __device__ __forceinline__ void async_load_b128(unsigned ldsAddr,
                                                       const void* gaddr) {
  asm volatile("global_load_async_to_lds_b128 %0, %1, off"
               :: "v"(ldsAddr), "v"(gaddr)
               : "memory");
}
static __device__ __forceinline__ void wait_asynccnt0() {
  asm volatile("s_wait_asynccnt 0" ::: "memory");
}
// Low 32 bits of a generic pointer to LDS == wave-relative LDS byte offset
// (ISA 10.2 aperture address calculation).
static __device__ __forceinline__ unsigned lds_off(const void* p) {
  return (unsigned)(unsigned long long)p;
}

// ---------------------------------------------------------------------------
// fp32 -> bf16 elementwise conversion, 4 elements per thread (grid-stride)
// ---------------------------------------------------------------------------
__global__ void cvt_f32_bf16(const float* __restrict__ in,
                             __bf16* __restrict__ out, int n4) {
  int i = blockIdx.x * blockDim.x + threadIdx.x;
  int stride = gridDim.x * blockDim.x;
  const float4* in4 = (const float4*)in;
  v4bf* out4 = (v4bf*)out;
  for (; i < n4; i += stride) {
    float4 f = in4[i];
    v4bf o;
    o[0] = (__bf16)f.x; o[1] = (__bf16)f.y;
    o[2] = (__bf16)f.z; o[3] = (__bf16)f.w;
    out4[i] = o;
  }
}

// ---------------------------------------------------------------------------
// Tiled WMMA GEMM: C[m,n] = sum_k A[m,k] * Bw[n,k]   (i.e. A @ Bw^T)
// A: [MTOT, DM] bf16 row-major.  Bw: [DM, DM] bf16 row-major (torch Linear W).
// Workgroup = 256 threads (8 waves), 128x128 output tile, K-step 32.
// Tiles staged with async global->LDS DMA, double-buffered, one barrier
// per K-step (s_wait_asynccnt 0 before the barrier).
// MODE 0: bf16 out, head-split [b,h,t,hd]
// MODE 1: bf16 out, head-split transposed [b,h,d,t]   (for V)
// MODE 2: fp32 out, row-major [m,n]                   (final projection)
// ---------------------------------------------------------------------------
template <int MODE>
__launch_bounds__(256)
__global__ void gemm_bf16_k(const __bf16* __restrict__ A,
                            const __bf16* __restrict__ Bw,
                            void* __restrict__ Cout) {
  __shared__ __bf16 sA[2][128 * 40]; // 128 rows x 32 cols, padded to 40
  __shared__ __bf16 sB[2][128 * 40];

  const int tid  = threadIdx.x;
  const int lane = tid & 31;
  const int wid  = tid >> 5;
  const int hw   = lane >> 4;   // half-wave select
  const int lr   = lane & 15;

  const int m0 = blockIdx.y * 128;
  const int n0 = blockIdx.x * 128;
  const int wm = (wid & 1) * 64;  // wave M offset (2 groups of 64)
  const int wn = (wid >> 1) * 32; // wave N offset (4 groups of 32)

  v8f acc[4][2];
#pragma unroll
  for (int i = 0; i < 4; ++i)
#pragma unroll
    for (int j = 0; j < 2; ++j)
#pragma unroll
      for (int e = 0; e < 8; ++e) acc[i][j][e] = 0.f;

  // cooperative staging: each thread DMAs one 32B row-chunk of A and of B
  const int ldr = tid >> 1;        // 0..127
  const int ldc = (tid & 1) * 16;  // 0 or 16
  const __bf16* gA = A  + (size_t)(m0 + ldr) * DM + ldc;
  const __bf16* gB = Bw + (size_t)(n0 + ldr) * DM + ldc;
  const int sOff = ldr * 40 + ldc;
  const unsigned lA0 = lds_off(sA[0] + sOff), lA1 = lds_off(sA[1] + sOff);
  const unsigned lB0 = lds_off(sB[0] + sOff), lB1 = lds_off(sB[1] + sOff);

  const int NK = DM / 32; // 64 K-steps

  // prologue: DMA tile 0 into buffer 0
  async_load_b128(lA0,      gA);
  async_load_b128(lA0 + 16, gA + 8);
  async_load_b128(lB0,      gB);
  async_load_b128(lB0 + 16, gB + 8);
  wait_asynccnt0();
  __syncthreads();

  for (int ks = 0; ks < NK; ++ks) {
    const int cur = ks & 1;
    const bool more = (ks + 1 < NK);

    // kick off next tile's DMA into the alternate buffer
    if (more) {
      const int kN = (ks + 1) * 32;
      const unsigned nA = cur ? lA0 : lA1;
      const unsigned nB = cur ? lB0 : lB1;
      async_load_b128(nA,      gA + kN);
      async_load_b128(nA + 16, gA + kN + 8);
      async_load_b128(nB,      gB + kN);
      async_load_b128(nB + 16, gB + kN + 8);
    }

    // B fragments (32x16): lanes 0-15 hold K 0..15 of column lr,
    // lanes 16-31 hold K 16..31 -> 32 contiguous bytes at col hw*16.
    v16bf bf0, bf1;
    {
      const __bf16* p = sB[cur] + (wn + lr) * 40 + hw * 16;
      bf0 = cat8(*(const v8bf*)p, *(const v8bf*)(p + 8));
      p = sB[cur] + (wn + 16 + lr) * 40 + hw * 16;
      bf1 = cat8(*(const v8bf*)p, *(const v8bf*)(p + 8));
    }
    // A fragments (16x32): lane row = lr; lanes 0-15 K 0..15,
    // lanes 16-31 K 16..31 -> 32 contiguous bytes at col hw*16.
    v16bf af[4];
#pragma unroll
    for (int mt = 0; mt < 4; ++mt) {
      const __bf16* p = sA[cur] + (wm + mt * 16 + lr) * 40 + hw * 16;
      af[mt] = cat8(*(const v8bf*)p, *(const v8bf*)(p + 8));
    }
#pragma unroll
    for (int mt = 0; mt < 4; ++mt) {
      acc[mt][0] = wmma_bf16(af[mt], bf0, acc[mt][0]);
      acc[mt][1] = wmma_bf16(af[mt], bf1, acc[mt][1]);
    }

    // next tile's DMA must have landed before anyone reads it
    if (more) wait_asynccnt0();
    __syncthreads();
  }

  // epilogue: C layout -> VGPR j, lanes 0-15: (M=j, N=lane); 16-31: (M=j+8)
#pragma unroll
  for (int mt = 0; mt < 4; ++mt)
#pragma unroll
    for (int nt = 0; nt < 2; ++nt)
#pragma unroll
      for (int j = 0; j < 8; ++j) {
        int m = m0 + wm + mt * 16 + j + hw * 8;
        int n = n0 + wn + nt * 16 + lr;
        float v = acc[mt][nt][j];
        if (MODE == 2) {
          ((float*)Cout)[(size_t)m * DM + n] = v;
        } else {
          int b = m >> 11, t = m & (SEQ - 1);
          int h = n >> 7,  d = n & (HD - 1);
          __bf16 bv = (__bf16)v;
          if (MODE == 0)
            ((__bf16*)Cout)[(((size_t)(b * NH + h)) * SEQ + t) * HD + d] = bv;
          else
            ((__bf16*)Cout)[(((size_t)(b * NH + h)) * HD + d) * SEQ + t] = bv;
        }
      }
}

// ---------------------------------------------------------------------------
// Flash attention (causal). Computes S^T = K * Q^T per 16-query x 32-key
// block so that P = exp(S^T - m) lands directly in WMMA A-fragment layout
// (keys along VGPRs, query columns along lanes): no LDS transpose needed.
// Q,K: [b,h,t,hd] bf16.  Vt: [b,h,d,t] bf16.  Ctx out: [b,t,(h,d)] bf16.
// Workgroup = 256 threads (8 waves); wave owns 16 query rows; block = 128.
// ---------------------------------------------------------------------------
__launch_bounds__(256)
__global__ void flash_attn_k(const __bf16* __restrict__ Q,
                             const __bf16* __restrict__ Kc,
                             const __bf16* __restrict__ Vt,
                             __bf16* __restrict__ Ctx) {
  const int lane = threadIdx.x & 31;
  const int wid  = threadIdx.x >> 5;
  const int hw   = lane >> 4;
  const int lr   = lane & 15;
  const int bh   = blockIdx.y;                  // b*NH + h
  const int qw0  = blockIdx.x * 128 + wid * 16; // first query row of this wave
  const int qcol = qw0 + lr;                    // this lane's query column

  const __bf16* qbase = Q  + ((size_t)bh * SEQ + qw0) * HD;
  const __bf16* kbase = Kc + (size_t)bh * SEQ * HD;
  const __bf16* vbase = Vt + (size_t)bh * (size_t)HD * SEQ;

  // Q^T B-fragments, one per 32-wide d-chunk, reused across all key blocks.
  v16bf qf[4];
#pragma unroll
  for (int dc = 0; dc < 4; ++dc) {
    const __bf16* p = qbase + (size_t)lr * HD + dc * 32 + hw * 16;
    qf[dc] = cat8(*(const v8bf*)p, *(const v8bf*)(p + 8));
  }

  v8f ctx[8];
#pragma unroll
  for (int nt = 0; nt < 8; ++nt)
#pragma unroll
    for (int e = 0; e < 8; ++e) ctx[nt][e] = 0.f;

  float m_i = -1e30f, l_i = 0.f;
  const float sc = 0.08838834764831845f; // 1/sqrt(128)
  const int kend = qw0 + 16;             // causal bound for this wave

  for (int k0 = 0; k0 < kend; k0 += 32) {
    // ---- batched K A-fragment loads (one big clause, one wait) ----
    v16bf ka[8]; // [0..3]=key tile0 per d-chunk, [4..7]=key tile1
#pragma unroll
    for (int dc = 0; dc < 4; ++dc) {
      const __bf16* p0 = kbase + (size_t)(k0 + lr) * HD + dc * 32 + hw * 16;
      ka[dc] = cat8(*(const v8bf*)p0, *(const v8bf*)(p0 + 8));
      const __bf16* p1 = p0 + (size_t)16 * HD;
      ka[4 + dc] = cat8(*(const v8bf*)p1, *(const v8bf*)(p1 + 8));
    }

    // ---- S^T WMMAs, split accumulation chains (depth 2 instead of 4) ----
    v8f s0a, s0b, s1a, s1b;
#pragma unroll
    for (int e = 0; e < 8; ++e) { s0a[e] = 0.f; s0b[e] = 0.f;
                                  s1a[e] = 0.f; s1b[e] = 0.f; }
    s0a = wmma_bf16(ka[0], qf[0], s0a);
    s1a = wmma_bf16(ka[4], qf[0], s1a);
    s0b = wmma_bf16(ka[1], qf[1], s0b);
    s1b = wmma_bf16(ka[5], qf[1], s1b);
    s0a = wmma_bf16(ka[2], qf[2], s0a);
    s1a = wmma_bf16(ka[6], qf[2], s1a);
    s0b = wmma_bf16(ka[3], qf[3], s0b);
    s1b = wmma_bf16(ka[7], qf[3], s1b);

    // ---- issue V B-fragment loads now; latency hides under softmax ----
    v16bf vf[8];
#pragma unroll
    for (int nt = 0; nt < 8; ++nt) {
      const __bf16* pv = vbase + (size_t)(nt * 16 + lr) * SEQ + k0 + hw * 8;
      vf[nt] = cat8(*(const v8bf*)pv, *(const v8bf*)(pv + 16));
    }
    // prefetch next key block (global_prefetch_b8)
    if (k0 + 32 < kend) {
      __builtin_prefetch(kbase + (size_t)(k0 + 32 + lr) * HD, 0, 3);
      __builtin_prefetch(kbase + (size_t)(k0 + 48 + lr) * HD, 0, 3);
    }

    v8f s0, s1;
#pragma unroll
    for (int e = 0; e < 8; ++e) { s0[e] = s0a[e] + s0b[e];
                                  s1[e] = s1a[e] + s1b[e]; }

    // scale + causal mask (row M = key = k0 + j + hw*8 [+16], col N = qcol)
    float mloc = -1e30f;
#pragma unroll
    for (int j = 0; j < 8; ++j) {
      int key0 = k0 + j + hw * 8;
      int key1 = key0 + 16;
      float v0 = (key0 <= qcol) ? s0[j] * sc : -1e30f;
      float v1 = (key1 <= qcol) ? s1[j] * sc : -1e30f;
      s0[j] = v0; s1[j] = v1;
      mloc = fmaxf(mloc, fmaxf(v0, v1));
    }
    // merge the two half-wave key groups of this query column
    mloc = fmaxf(mloc, __shfl_xor(mloc, 16, 32));
    float mnew  = fmaxf(m_i, mloc);
    float alpha = __expf(m_i - mnew);
    float lsum  = 0.f;

    // P tile: C layout of S^T == A-fragment layout for P (16 q x 32 keys)
    v16bf pfrag;
#pragma unroll
    for (int j = 0; j < 8; ++j) {
      float e0 = __expf(s0[j] - mnew);
      float e1 = __expf(s1[j] - mnew);
      lsum += e0 + e1;
      pfrag[j]     = (__bf16)e0;
      pfrag[j + 8] = (__bf16)e1;
    }
    lsum += __shfl_xor(lsum, 16, 32);
    l_i = l_i * alpha + lsum;
    m_i = mnew;

    // rescale ctx: factor is per query ROW of the ctx tile (M = j + hw*8);
    // that row's alpha lives in lane (j + hw*8) -> ds_bpermute broadcast.
#pragma unroll
    for (int j = 0; j < 8; ++j) {
      float aj = __shfl(alpha, j + hw * 8, 32);
#pragma unroll
      for (int nt = 0; nt < 8; ++nt) ctx[nt][j] *= aj;
    }

    // ---- ctx += P * V : 8 independent WMMAs, no waits in between ----
#pragma unroll
    for (int nt = 0; nt < 8; ++nt)
      ctx[nt] = wmma_bf16(pfrag, vf[nt], ctx[nt]);
  }

  // normalize by l and store ctx as bf16 in [b, t, h*HD + d]
  float rinv = 1.f / l_i;
  const int b = bh >> 4, h = bh & (NH - 1);
  __bf16* cb = Ctx + (size_t)b * SEQ * DM + (size_t)h * HD;
#pragma unroll
  for (int j = 0; j < 8; ++j) {
    float rj = __shfl(rinv, j + hw * 8, 32);
    int q = qw0 + j + hw * 8;
#pragma unroll
    for (int nt = 0; nt < 8; ++nt) {
      int d = nt * 16 + lr;
      cb[(size_t)q * DM + d] = (__bf16)(ctx[nt][j] * rj);
    }
  }
}

// ---------------------------------------------------------------------------
// Host-side launch sequence
// ---------------------------------------------------------------------------
extern "C" void kernel_launch(void* const* d_in, const int* in_sizes, int n_in,
                              void* d_out, int out_size, void* d_ws,
                              size_t ws_size, hipStream_t stream) {
  (void)in_sizes; (void)n_in; (void)out_size; (void)ws_size;
  const float* x  = (const float*)d_in[0];
  const float* Wq = (const float*)d_in[1];
  const float* Wk = (const float*)d_in[2];
  const float* Wv = (const float*)d_in[3];
  const float* Wo = (const float*)d_in[4];

  char* ws = (char*)d_ws;
  const size_t MB = 1ull << 20;
  __bf16* xb  = (__bf16*)(ws + 0);         // 32 MB, later reused as ctx
  __bf16* wqb = (__bf16*)(ws + 32 * MB);   // 8 MB each
  __bf16* wkb = (__bf16*)(ws + 40 * MB);
  __bf16* wvb = (__bf16*)(ws + 48 * MB);
  __bf16* wob = (__bf16*)(ws + 56 * MB);
  __bf16* qb  = (__bf16*)(ws + 64 * MB);   // [b,h,t,hd] 32 MB
  __bf16* kb  = (__bf16*)(ws + 96 * MB);   // [b,h,t,hd] 32 MB
  __bf16* vtb = (__bf16*)(ws + 128 * MB);  // [b,h,d,t]  32 MB
  __bf16* ctx = xb;                         // xb dead after V GEMM

  const int XN4 = (MTOT * DM) / 4; // 4,194,304 float4s
  const int WN4 = (DM * DM) / 4;   // 1,048,576 float4s

  cvt_f32_bf16<<<(XN4 + 1023) / 1024, 256, 0, stream>>>(x,  xb,  XN4);
  cvt_f32_bf16<<<(WN4 + 1023) / 1024, 256, 0, stream>>>(Wq, wqb, WN4);
  cvt_f32_bf16<<<(WN4 + 1023) / 1024, 256, 0, stream>>>(Wk, wkb, WN4);
  cvt_f32_bf16<<<(WN4 + 1023) / 1024, 256, 0, stream>>>(Wv, wvb, WN4);
  cvt_f32_bf16<<<(WN4 + 1023) / 1024, 256, 0, stream>>>(Wo, wob, WN4);

  dim3 gg(DM / 128, MTOT / 128); // (16, 64)
  gemm_bf16_k<0><<<gg, 256, 0, stream>>>(xb, wqb, qb);
  gemm_bf16_k<0><<<gg, 256, 0, stream>>>(xb, wkb, kb);
  gemm_bf16_k<1><<<gg, 256, 0, stream>>>(xb, wvb, vtb);

  dim3 ga(SEQ / 128, BATCH * NH); // (16, 64)
  flash_attn_k<<<ga, 256, 0, stream>>>(qb, kb, vtb, ctx);

  gemm_bf16_k<2><<<gg, 256, 0, stream>>>(ctx, wob, (float*)d_out);
}
